// ResNet_618475291155
// MI455X (gfx1250) — matrix operand order repaired and checked by reading, not compiled, not run on hardware
//
#include <hip/hip_runtime.h>

typedef __attribute__((ext_vector_type(16))) __bf16 v16bf;
typedef __attribute__((ext_vector_type(8)))  float  v8f;
typedef int v4i __attribute__((vector_size(16)));

#if defined(__HIP_DEVICE_COMPILE__) && __has_builtin(__builtin_amdgcn_global_load_async_to_lds_b128)
#define HAVE_ASYNC_LDS 1
typedef __attribute__((address_space(1))) v4i* gas_v4i_p;
typedef __attribute__((address_space(3))) v4i* las_v4i_p;
#else
#define HAVE_ASYNC_LDS 0
#endif

union FragU { unsigned int u[8]; v16bf v; };

__device__ inline unsigned short f2bf(float f) {
    unsigned u = __float_as_uint(f);
    unsigned r = u + 0x7FFFu + ((u >> 16) & 1u);
    return (unsigned short)(r >> 16);
}

// A fragment: 16x32 bf16, rows (M) in lanes, contiguous k = 16*(lane/16) + s (matches B).
__device__ inline v16bf load_frag_A(const unsigned short* base, int stride, int lane) {
    int row = lane & 15;
    int kb  = (lane & 16) ? 16 : 0;
    const unsigned short* p = base + row * stride;
    FragU f;
#pragma unroll
    for (int v = 0; v < 4; ++v) f.u[v] = *(const unsigned int*)(p + kb + 2 * v);
#pragma unroll
    for (int v = 4; v < 8; ++v) f.u[v] = *(const unsigned int*)(p + kb + 2 * v);
    return f.v;
}

// B fragment: 32x16 bf16 (KxN), stored [n][k], cols in lanes.
__device__ inline v16bf load_frag_B(const unsigned short* base, int stride, int lane) {
    int col = lane & 15;
    int kb  = (lane & 16) ? 16 : 0;
    const unsigned short* p = base + col * stride + kb;
    FragU f;
#pragma unroll
    for (int v = 0; v < 8; ++v) f.u[v] = *(const unsigned int*)(p + 2 * v);
    return f.v;
}

// ---------------- fp32 weights -> bf16, K padded to Kp (zeros) ----------------
__global__ void w2bf_k(const float* __restrict__ W, unsigned short* __restrict__ O,
                       int Cout, int K, int Kp)
{
    long i = (long)blockIdx.x * blockDim.x + threadIdx.x;
    long tot = (long)Cout * Kp;
    if (i >= tot) return;
    int c = (int)(i / Kp), k = (int)(i % Kp);
    O[i] = (k < K) ? f2bf(W[(long)c * K + k]) : (unsigned short)0;
}

// ---------------- Implicit-GEMM conv + BN (+residual)(+ReLU) ----------------
// C[M=Cout, N=B*OH*OW], K = Cin*KH*KW (padded to Kp in Wbf). A = weights, B = im2col.
#define BM 128
#define BN 64
#define BK 32

template<int TKH, int TKW>
__global__ __launch_bounds__(256)
void conv_gemm(const float* __restrict__ X, const unsigned short* __restrict__ Wbf,
               const float* __restrict__ bg, const float* __restrict__ bb,
               const float* __restrict__ bm, const float* __restrict__ bv,
               const float* __restrict__ R, float* __restrict__ Y,
               int B, int Cin, int H, int Win, int Cout,
               int stride, int pad, int OH, int OW, int Kp, int relu)
{
    __shared__ __attribute__((aligned(16))) unsigned short As[BM * BK];
    __shared__ __attribute__((aligned(16))) unsigned short Bs[BN * BK];

    const long Ntot = (long)B * OH * OW;
    const int  K    = Cin * TKH * TKW;
    const int  n0   = blockIdx.x * BN;
    const int  m0   = blockIdx.y * BM;
    const int  tid  = threadIdx.x;
    const int  lane = tid & 31;
    const int  wave = tid >> 5;
    const int  wm   = wave & 3;   // 4 waves over M (32 rows each)
    const int  wn   = wave >> 2;  // 2 waves over N (32 cols each)
    const long HWl  = (long)H * Win;

    v8f acc[2][2] = {};

    // A staging coords: unconditional (row clamped; Kp zero-padded)
    const int am  = tid >> 1;               // 0..127 row
    const int ah  = (tid & 1) * 16;         // k half
    const int gm  = m0 + am;
    const int gmc = gm < Cout ? gm : Cout - 1;
    const unsigned short* wrow = Wbf + (long)gmc * Kp + ah;

    // B staging coords (n fixed per thread)
    const int bnn = tid >> 2;               // 0..63: n within tile
    const int bq  = tid & 3;                // k-quarter (8 elems)
    const long nn = (long)n0 + bnn;
    const bool nok = nn < Ntot;
    const long nnc = nok ? nn : 0;
    const int ob = (int)(nnc / ((long)OH * OW));
    const int sp0 = (int)(nnc % ((long)OH * OW));
    const int oh = sp0 / OW, ow = sp0 % OW;
    const float nokf = nok ? 1.f : 0.f;
    const long bbase = (long)ob * Cin * HWl + (long)(oh * stride) * Win + (long)(ow * stride);

    for (int k0 = 0; k0 < Kp; k0 += BK) {
        { // A: contiguous bf16 global -> LDS (async-to-LDS when available)
            const unsigned short* gp = wrow + k0;
#if HAVE_ASYNC_LDS
            __builtin_amdgcn_global_load_async_to_lds_b128(
                (gas_v4i_p)gp, (las_v4i_p)&As[am * BK + ah], 0, 0);
            __builtin_amdgcn_global_load_async_to_lds_b128(
                (gas_v4i_p)(gp + 8), (las_v4i_p)&As[am * BK + ah + 8], 0, 0);
#else
            uint4 a0 = ((const uint4*)gp)[0];
            uint4 a1 = ((const uint4*)gp)[1];
            *(uint4*)&As[am * BK + ah]     = a0;
            *(uint4*)&As[am * BK + ah + 8] = a1;
#endif
        }
        { // B: gather 8, pack, one 16B LDS store
            unsigned pk[4];
            if (TKH == 1 && TKW == 1) {
#pragma unroll
                for (int i2 = 0; i2 < 8; i2 += 2) {
                    int kk = k0 + bq * 8 + i2;
                    float v0 = X[bbase + (long)kk * HWl] * nokf;
                    float v1 = X[bbase + (long)(kk + 1) * HWl] * nokf;
                    pk[i2 >> 1] = (unsigned)f2bf(v0) | ((unsigned)f2bf(v1) << 16);
                }
            } else {
                const int KHW = TKH * TKW;
                int kk0 = k0 + bq * 8;
                int ci = kk0 / KHW;
                int rr = kk0 - ci * KHW;           // incremental (ci, kh, kw) decode
                unsigned short us[8];
#pragma unroll
                for (int i2 = 0; i2 < 8; ++i2) {
                    int kh = rr / TKW, kw2 = rr - kh * TKW;
                    int ih = oh * stride - pad + kh;
                    int iw = ow * stride - pad + kw2;
                    int kk = kk0 + i2;
                    bool ok = nok && (kk < K) && (ih >= 0) && (ih < H) && (iw >= 0) && (iw < Win);
                    long idx = ok ? (((long)ob * Cin + ci) * HWl + (long)ih * Win + iw) : 0;
                    float v = X[idx];
                    us[i2] = f2bf(ok ? v : 0.f);
                    rr += 1;
                    if (rr == KHW) { rr = 0; ++ci; }
                }
#pragma unroll
                for (int i2 = 0; i2 < 4; ++i2)
                    pk[i2] = (unsigned)us[2 * i2] | ((unsigned)us[2 * i2 + 1] << 16);
            }
            uint4 bvv; bvv.x = pk[0]; bvv.y = pk[1]; bvv.z = pk[2]; bvv.w = pk[3];
            *(uint4*)&Bs[bnn * BK + bq * 8] = bvv;
        }
#if HAVE_ASYNC_LDS
#if __has_builtin(__builtin_amdgcn_s_wait_asynccnt)
        __builtin_amdgcn_s_wait_asynccnt(0);
#else
        asm volatile("s_wait_asynccnt 0x0" ::: "memory");
#endif
#endif
        __syncthreads();
#pragma unroll
        for (int i = 0; i < 2; ++i) {
            v16bf fa = load_frag_A(&As[(wm * 32 + i * 16) * BK], BK, lane);
#pragma unroll
            for (int j = 0; j < 2; ++j) {
                v16bf fb = load_frag_B(&Bs[(wn * 32 + j * 16) * BK], BK, lane);
                acc[i][j] = __builtin_amdgcn_wmma_f32_16x16x32_bf16(
                    false, fa, false, fb, (short)0, acc[i][j], false, false);
            }
        }
        __syncthreads();
    }

    // Epilogue: BN fold + optional residual + optional ReLU, NCHW scatter
    const int col  = lane & 15;
    const int rofs = (lane & 16) ? 8 : 0;
    const long OHW = (long)OH * OW;
    long nb[2]; int bo[2], so[2];
#pragma unroll
    for (int j = 0; j < 2; ++j) {
        nb[j] = (long)n0 + wn * 32 + j * 16 + col;
        long nc = nb[j] < Ntot ? nb[j] : 0;
        bo[j] = (int)(nc / OHW); so[j] = (int)(nc - (long)bo[j] * OHW);
    }
#pragma unroll
    for (int i = 0; i < 2; ++i) {
#pragma unroll
        for (int r = 0; r < 8; ++r) {
            int m = m0 + wm * 32 + i * 16 + rofs + r;
            bool mok = m < Cout;
            int mc = mok ? m : 0;
            float sc = bg[mc] * rsqrtf(bv[mc] + 1e-5f);
            float sh = bb[mc] - bm[mc] * sc;
#pragma unroll
            for (int j = 0; j < 2; ++j) {
                if (!mok || nb[j] >= Ntot) continue;
                float v = acc[i][j][r] * sc + sh;
                long oidx = ((long)bo[j] * Cout + m) * OHW + so[j];
                if (R) v += R[oidx];
                if (relu) v = fmaxf(v, 0.f);
                Y[oidx] = v;
            }
        }
    }
}

// ---------------- MaxPool 3x3 s2 p1 ----------------
__global__ void maxpool_k(const float* __restrict__ X, float* __restrict__ Y,
                          int B, int C, int H, int W, int OH, int OW)
{
    long i = (long)blockIdx.x * blockDim.x + threadIdx.x;
    long tot = (long)B * C * OH * OW;
    if (i >= tot) return;
    int ow = (int)(i % OW); long t = i / OW;
    int oh = (int)(t % OH); t /= OH;
    int c  = (int)(t % C);  int b = (int)(t / C);
    float m = -3.402823466e38f;
    for (int kh = 0; kh < 3; ++kh)
        for (int kw = 0; kw < 3; ++kw) {
            int ih = oh * 2 - 1 + kh, iw = ow * 2 - 1 + kw;
            if (ih >= 0 && ih < H && iw >= 0 && iw < W)
                m = fmaxf(m, X[(((long)b * C + c) * H + ih) * W + iw]);
        }
    Y[i] = m;
}

// ---------------- content [16,2048,7,7] -> xs bf16 [392,16,256] ----------------
__global__ void make_xs_k(const float* __restrict__ content, unsigned short* __restrict__ xs)
{
    int i = blockIdx.x * blockDim.x + threadIdx.x;
    if (i >= 392 * 16 * 256) return;
    int d = i & 255; int t = i >> 8; int b = t & 15; int tt = t >> 4;  // tt = step
    int ii = tt >> 3; int j = tt & 7;                                  // t = i*8+j
    float v = content[((long)b * 2048 + j * 256 + d) * 49 + ii];
    xs[i] = f2bf(v);
}

// ---------------- Persistent GRU scan: 392 steps, one workgroup, WMMA ----------------
__global__ __launch_bounds__(512)
void gru_scan(const unsigned short* __restrict__ xs,
              const unsigned short* __restrict__ wihb, const unsigned short* __restrict__ whhb,
              const float* __restrict__ bih, const float* __restrict__ bhh,
              float* __restrict__ hs)
{
    __shared__ __attribute__((aligned(16))) float Gi[16 * 768];
    __shared__ __attribute__((aligned(16))) float Gh[16 * 768];
    __shared__ __attribute__((aligned(16))) float Hst[16 * 256];
    __shared__ __attribute__((aligned(16))) unsigned short Hbf[16 * 256];
    __shared__ __attribute__((aligned(16))) unsigned short Xbf[16 * 256];

    const int tid = threadIdx.x;
    const int lane = tid & 31, wave = tid >> 5;  // 16 waves
    for (int i = tid; i < 16 * 256; i += 512) Hst[i] = 0.f;
    for (int i = tid; i < 512; i += 512) ((uint4*)Hbf)[i] = uint4{0, 0, 0, 0};
    __syncthreads();

    for (int t = 0; t < 392; ++t) {
        {
            const uint4* src = (const uint4*)(xs + (long)t * 4096);
            ((uint4*)Xbf)[tid] = src[tid];   // 512 * 16B = 8KB
        }
        __syncthreads();

        // 96 WMMA tile-jobs: 48 gi tiles (N=768) + 48 gh tiles; 6 per wave
        for (int jj = 0; jj < 6; ++jj) {
            int job = wave * 6 + jj;
            int isgi = (job < 48);
            int nt = isgi ? job : job - 48;
            const unsigned short* Wp = isgi ? wihb : whhb;
            const unsigned short* Ap = isgi ? Xbf : Hbf;
            const int col = lane & 15;
            const int kb = (lane & 16) ? 16 : 0;
            const unsigned short* wrow = Wp + (long)(nt * 16 + col) * 256 + kb;
            v8f acc = {};
            for (int k0 = 0; k0 < 256; k0 += 32) {
                v16bf fa = load_frag_A(Ap + k0, 256, lane);
                FragU fb;
                const uint4* p = (const uint4*)(wrow + k0);
                *(uint4*)&fb.u[0] = p[0];
                *(uint4*)&fb.u[4] = p[1];
                acc = __builtin_amdgcn_wmma_f32_16x16x32_bf16(
                    false, fa, false, fb.v, (short)0, acc, false, false);
            }
            float* G = isgi ? Gi : Gh;
            const float* bias = isgi ? bih : bhh;
            int rofs = (lane & 16) ? 8 : 0;
            int n = nt * 16 + col;
            float bs = bias[n];
#pragma unroll
            for (int r = 0; r < 8; ++r) G[(rofs + r) * 768 + n] = acc[r] + bs;
        }
        __syncthreads();

        // gate math + state update
        for (int i = tid; i < 4096; i += 512) {
            int m = i >> 8; int d = i & 255;
            float ir = Gi[m * 768 + d], iz = Gi[m * 768 + 256 + d], inn = Gi[m * 768 + 512 + d];
            float hr = Gh[m * 768 + d], hz = Gh[m * 768 + 256 + d], hn  = Gh[m * 768 + 512 + d];
            float r = 1.f / (1.f + __expf(-(ir + hr)));
            float z = 1.f / (1.f + __expf(-(iz + hz)));
            float n = tanhf(inn + r * hn);
            float h2 = (1.f - z) * n + z * Hst[i];
            Hst[i] = h2; Hbf[i] = f2bf(h2);
            hs[(long)t * 4096 + i] = h2;
        }
        __syncthreads();
    }
}

// ---------------- mu head + softmax + argmax/logp, one block per (i,b) ----------------
__global__ __launch_bounds__(64)
void mu_head(const float* __restrict__ hs, const float* __restrict__ muw,
             const float* __restrict__ mub, float* __restrict__ actionT,
             float* __restrict__ logpT)
{
    int ib = blockIdx.x;            // i*16 + b
    int i = ib >> 4, b = ib & 15;
    __shared__ float hsum[256];
    __shared__ float mu[53];
    int tid = threadIdx.x;
    for (int d = tid; d < 256; d += 64) {
        float s = 0.f;
        for (int j = 0; j < 8; ++j) s += hs[((long)(i * 8 + j) * 16 + b) * 256 + d];
        hsum[d] = s;
    }
    __syncthreads();
    for (int a = tid; a < 53; a += 64) {
        float s = 8.f * mub[a];
        for (int d = 0; d < 256; ++d) s += hsum[d] * muw[a * 256 + d];
        mu[a] = s;
    }
    __syncthreads();
    if (tid == 0) {
        float mx = -3.402823466e38f;
        for (int a = 0; a < 53; ++a) mx = fmaxf(mx, mu[a]);
        float den = 0.f;
        for (int a = 0; a < 53; ++a) den += __expf(mu[a] - mx);
        float best = -3.402823466e38f; int bi = 0;
        for (int a = 0; a < 53; ++a) {
            float y = __expf(mu[a] - mx) / den;
            if (y > best) { best = y; bi = a; }
        }
        best = fminf(fmaxf(best, 1e-20f), 1e20f);
        actionT[b * 49 + i] = (float)bi;
        logpT[b * 49 + i]   = __logf(best);
    }
}

// ---------------- attention pooling + BN + final linears, one block per b ----------------
__global__ __launch_bounds__(256)
void final_head(const float* __restrict__ hs, const float* __restrict__ content,
                const float* __restrict__ actionT,
                const float* bng, const float* bnb, const float* bnm, const float* bnv,
                const float* lastw, const float* lastb,
                const float* disw, const float* disb,
                const float* clsw, const float* clsb,
                float* __restrict__ out_cls, float* __restrict__ out_fc,
                float* __restrict__ out_cc)
{
    int b = blockIdx.x;
    __shared__ float att[49];
    __shared__ float fc[2048];
    __shared__ float cf[1024];
    int tid = threadIdx.x;
    if (tid == 0) {
        float mx = -3.402823466e38f;
        for (int s = 0; s < 49; ++s) mx = fmaxf(mx, actionT[b * 49 + s] * 20.f);
        float den = 0.f;
        for (int s = 0; s < 49; ++s) { att[s] = __expf(actionT[b * 49 + s] * 20.f - mx); den += att[s]; }
        for (int s = 0; s < 49; ++s) att[s] /= den;
    }
    __syncthreads();
    for (int d = tid; d < 2048; d += 256) {
        int j = d >> 8; int dd = d & 255;
        float f = 0.f;
        for (int s = 0; s < 49; ++s) f += hs[((long)(s * 8 + j) * 16 + b) * 256 + dd] * att[s];
        float p = 0.f;
        for (int s = 0; s < 49; ++s) p += content[((long)b * 2048 + d) * 49 + s];
        p *= (1.f / 49.f);
        float sc = bng[d] * rsqrtf(bnv[d] + 1e-5f);
        float sh = bnb[d] - bnm[d] * sc;
        float v = (f + p) * sc + sh;
        fc[d] = v;
        out_fc[b * 2048 + d] = v;
    }
    __syncthreads();
    for (int a = tid; a < 130; a += 256) {
        float s = lastb[a];
        for (int d = 0; d < 2048; ++d) s += fc[d] * lastw[a * 2048 + d];
        out_cls[b * 130 + a] = s;
    }
    for (int a = tid; a < 1024; a += 256) {
        float s = disb[a];
        for (int d = 0; d < 2048; ++d) s += fc[d] * disw[a * 2048 + d];
        cf[a] = fmaxf(s, 0.f);
    }
    __syncthreads();
    if (tid == 0) {
        float s = clsb[0];
        for (int d = 0; d < 1024; ++d) s += cf[d] * clsw[d];
        out_cc[b] = s;
    }
}

// ================================ host ================================
struct BlockP {
    const float *b1[4], *b2[4], *b3[4];   // [b,g,m,v] each (sorted keys)
    const float *c1, *c2, *c3;
    const float *db[4]; const float *dc;
    bool has_ds;
};

extern "C" void kernel_launch(void* const* d_in, const int* in_sizes, int n_in,
                              void* d_out, int out_size, void* d_ws, size_t ws_size,
                              hipStream_t stream)
{
    (void)in_sizes; (void)n_in; (void)out_size; (void)ws_size;
    int idx = 0;
    const float* x = (const float*)d_in[idx++];

    // params: JAX pytree order (dict keys sorted alphabetically)
    const float* bn_[4];  for (int q = 0; q < 4; ++q) bn_[q]  = (const float*)d_in[idx++]; // bn:  b,g,m,v
    const float* bn1_[4]; for (int q = 0; q < 4; ++q) bn1_[q] = (const float*)d_in[idx++]; // bn1: b,g,m,v
    const float* cls_b = (const float*)d_in[idx++]; const float* cls_w = (const float*)d_in[idx++];
    const float* conv1_w = (const float*)d_in[idx++];
    const float* dis_b = (const float*)d_in[idx++]; const float* dis_w = (const float*)d_in[idx++];
    const float* gru_bhh = (const float*)d_in[idx++]; const float* gru_bih = (const float*)d_in[idx++];
    const float* gru_whh = (const float*)d_in[idx++]; const float* gru_wih = (const float*)d_in[idx++];
    const float* last_b = (const float*)d_in[idx++]; const float* last_w = (const float*)d_in[idx++];
    const float* mu_b = (const float*)d_in[idx++];   const float* mu_w = (const float*)d_in[idx++];

    static const int blocks_[4] = {3, 4, 6, 3};
    static const int planes_[4] = {64, 128, 256, 512};
    static const int stride_[4] = {1, 2, 2, 2};
    BlockP blk[16]; int nb = 0;
    for (int st = 0; st < 4; ++st)
        for (int bi = 0; bi < blocks_[st]; ++bi) {
            BlockP& p = blk[nb++];
            for (int q = 0; q < 4; ++q) p.b1[q] = (const float*)d_in[idx++];
            for (int q = 0; q < 4; ++q) p.b2[q] = (const float*)d_in[idx++];
            for (int q = 0; q < 4; ++q) p.b3[q] = (const float*)d_in[idx++];
            p.c1 = (const float*)d_in[idx++];
            p.c2 = (const float*)d_in[idx++];
            p.c3 = (const float*)d_in[idx++];
            p.has_ds = (bi == 0);
            if (p.has_ds) {
                for (int q = 0; q < 4; ++q) p.db[q] = (const float*)d_in[idx++];
                p.dc = (const float*)d_in[idx++];
            } else { p.dc = nullptr; }
        }

    // workspace carve (floats, then ushort tail)
    const long ACT = 16L * 256 * 56 * 56;  // largest activation
    float* P0  = (float*)d_ws;
    float* P1  = P0 + ACT;
    float* T1  = P1 + ACT;
    float* T2  = T1 + 16L * 128 * 56 * 56;
    float* DSB = T2 + 16L * 64 * 56 * 56;
    float* HS  = DSB + ACT;                               // [392,16,256] fp32
    unsigned short* XS = (unsigned short*)(HS + 392L * 16 * 256);
    unsigned short* wcur = XS + 392L * 16 * 256;          // bf16 weight arena

    float* out = (float*)d_out;
    float* out_cls    = out;                 // [16,130]
    float* out_fc     = out + 2080;          // [16,2048]
    float* out_cc     = out + 34848;         // [16,1]
    float* out_action = out + 34864;         // [16,1,49]
    float* out_logp   = out + 35648;         // [16,1,49]

    auto conv = [&](const float* Xp, const float* Wp, const float* const bnp[4],
                    const float* Rp, float* Yp, int Cin, int H, int Cout,
                    int KH, int s, int pad, int relu) {
        int K = Cin * KH * KH;
        int Kp = (K + 31) & ~31;
        long wtot = (long)Cout * Kp;
        w2bf_k<<<(unsigned)((wtot + 255) / 256), 256, 0, stream>>>(Wp, wcur, Cout, K, Kp);
        int OH = (H + 2 * pad - KH) / s + 1;
        long N = 16L * OH * OH;
        dim3 g((unsigned)((N + BN - 1) / BN), (unsigned)((Cout + BM - 1) / BM));
        if (KH == 1)
            conv_gemm<1, 1><<<g, 256, 0, stream>>>(Xp, wcur, bnp[1], bnp[0], bnp[2], bnp[3], Rp, Yp,
                                                   16, Cin, H, H, Cout, s, pad, OH, OH, Kp, relu);
        else if (KH == 3)
            conv_gemm<3, 3><<<g, 256, 0, stream>>>(Xp, wcur, bnp[1], bnp[0], bnp[2], bnp[3], Rp, Yp,
                                                   16, Cin, H, H, Cout, s, pad, OH, OH, Kp, relu);
        else
            conv_gemm<7, 7><<<g, 256, 0, stream>>>(Xp, wcur, bnp[1], bnp[0], bnp[2], bnp[3], Rp, Yp,
                                                   16, Cin, H, H, Cout, s, pad, OH, OH, Kp, relu);
        wcur += wtot;
    };

    // conv1 7x7 s2 p3 + BN + ReLU: x[16,3,224,224] -> P0[16,64,112,112]
    conv(x, conv1_w, bn1_, nullptr, P0, 3, 224, 64, 7, 2, 3, 1);
    // maxpool -> P1 [16,64,56,56]
    {
        long tot = 16L * 64 * 56 * 56;
        maxpool_k<<<(unsigned)((tot + 255) / 256), 256, 0, stream>>>(P0, P1, 16, 64, 112, 112, 56, 56);
    }

    float* cur = P1; float* oth = P0;
    int inpl = 64, Hc = 56, ib = 0;
    for (int st = 0; st < 4; ++st) {
        int pl = planes_[st];
        for (int bi = 0; bi < blocks_[st]; ++bi, ++ib) {
            int s = (bi == 0) ? stride_[st] : 1;
            BlockP& bp = blk[ib];
            int Ho = (Hc - 1) / s + 1;
            conv(cur, bp.c1, bp.b1, nullptr, T1, inpl, Hc, pl, 1, 1, 0, 1);       // 1x1 + BN + ReLU
            conv(T1,  bp.c2, bp.b2, nullptr, T2, pl,   Hc, pl, 3, s, 1, 1);       // 3x3 s + BN + ReLU
            const float* resid = cur;
            if (bp.has_ds) {
                conv(cur, bp.dc, bp.db, nullptr, DSB, inpl, Hc, pl * 4, 1, s, 0, 0);
                resid = DSB;
            }
            conv(T2, bp.c3, bp.b3, resid, oth, pl, Ho, pl * 4, 1, 1, 0, 1);       // 1x1 + BN + add + ReLU
            float* t = cur; cur = oth; oth = t;
            inpl = pl * 4; Hc = Ho;
        }
    }
    const float* content = cur;  // [16,2048,7,7]

    make_xs_k<<<(392 * 4096 + 255) / 256, 256, 0, stream>>>(content, XS);

    // GRU weights -> bf16
    unsigned short* wihb = wcur; wcur += 768L * 256;
    unsigned short* whhb = wcur; wcur += 768L * 256;
    w2bf_k<<<(768 * 256 + 255) / 256, 256, 0, stream>>>(gru_wih, wihb, 768, 256, 256);
    w2bf_k<<<(768 * 256 + 255) / 256, 256, 0, stream>>>(gru_whh, whhb, 768, 256, 256);

    gru_scan<<<1, 512, 0, stream>>>(XS, wihb, whhb, gru_bih, gru_bhh, HS);
    mu_head<<<784, 64, 0, stream>>>(HS, mu_w, mu_b, out_action, out_logp);
    final_head<<<16, 256, 0, stream>>>(HS, content, out_action,
                                       bn_[1], bn_[0], bn_[2], bn_[3],
                                       last_w, last_b, dis_w, dis_b, cls_w, cls_b,
                                       out_cls, out_fc, out_cc);
}
